// TorchHawkes_21371757265066
// MI455X (gfx1250) — matrix operand order repaired and checked by reading, not compiled, not run on hardware
//
#include <hip/hip_runtime.h>
#include <hip/hip_bf16.h>
#include <math.h>

typedef _Float16 h8   __attribute__((ext_vector_type(8)));
typedef _Float16 v16h __attribute__((ext_vector_type(16)));
typedef float    v8f  __attribute__((ext_vector_type(8)));
typedef float    f4   __attribute__((ext_vector_type(4)));

#define KK   1024
#define NN   16384
#define NCH  64      // scan chunks
#define CHL  256     // chunk length (NCH*CHL == NN)

// ---------------- Phase 0: Mu0 = row-mean(obs)/10 + 1e-5 ----------------
__global__ __launch_bounds__(256)
void mu0_kernel(const float* __restrict__ obs, float* __restrict__ Mu0) {
  int j = blockIdx.x;
  const float* row = obs + (size_t)j * NN;
  float s = 0.f;
  for (int t = threadIdx.x; t < NN; t += 256) s += row[t];
  __shared__ float red[256];
  red[threadIdx.x] = s; __syncthreads();
  for (int st = 128; st > 0; st >>= 1) {
    if (threadIdx.x < st) red[threadIdx.x] += red[threadIdx.x + st];
    __syncthreads();
  }
  if (threadIdx.x == 0) Mu0[j] = red[0] * (1.0f / (float)NN) * 0.1f + 1e-5f;
}

// ---------------- Alpha f32 -> f16 ----------------
__global__ __launch_bounds__(256)
void a2h_kernel(const float* __restrict__ A, _Float16* __restrict__ AH) {
  int i = blockIdx.x * 256 + threadIdx.x;
  AH[i] = (_Float16)A[i];
}

// ---------------- Scan pass A: chunk-local end carries (S start = 0) ----------------
__global__ __launch_bounds__(256)
void scanA_kernel(const float* __restrict__ obs, const float* __restrict__ Beta,
                  float* __restrict__ E) {
  int gid = blockIdx.x * 256 + threadIdx.x;   // KK*NCH threads
  int j = gid & (KK - 1);
  int c = gid >> 10;
  float d = __expf(-Beta[j]);
  const float* row = obs + (size_t)j * NN + c * CHL;
  float S = 0.f;
  for (int t = 0; t < CHL; ++t) S = d * (S + row[t]);
  E[c * KK + j] = S;
}

// ---------------- Scan pass B: per-channel prefix over chunks ----------------
__global__ __launch_bounds__(256)
void scanP_kernel(const float* __restrict__ Beta, const float* __restrict__ E,
                  float* __restrict__ Astart) {
  int j = blockIdx.x * 256 + threadIdx.x;     // KK threads
  float dL = __expf(-Beta[j] * (float)CHL);   // decay^CHL exactly
  float A = 0.f;
  for (int c = 0; c < NCH; ++c) {
    Astart[c * KK + j] = A;
    A = dL * A + E[c * KK + j];
  }
}

// ---------------- Scan pass C: emit G[t,j] = beta_j * S_t (f16) ----------------
__global__ __launch_bounds__(256)
void scanB_kernel(const float* __restrict__ obs, const float* __restrict__ Beta,
                  const float* __restrict__ Astart, _Float16* __restrict__ G) {
  int gid = blockIdx.x * 256 + threadIdx.x;   // KK*NCH threads
  int j = gid & (KK - 1);
  int c = gid >> 10;
  float b = Beta[j];
  float d = __expf(-b);
  float S = Astart[c * KK + j];
  const float* row = obs + (size_t)j * NN;
  int t0 = c * CHL;
  for (int t = t0; t < t0 + CHL; ++t) {
    G[(size_t)t * KK + j] = (_Float16)(b * S);   // coalesced across j
    S = d * (S + row[t]);
  }
}

// ---------------- Fused WMMA GEMM + softplus + outputs + loglik ----------------
__device__ __attribute__((always_inline)) inline
float epilogue_tile(v8f d, int k, int t0,
                    const float* __restrict__ obs,
                    const float* __restrict__ Mu0,
                    float* __restrict__ lams0,
                    float* __restrict__ lam1) {
  float mu = Mu0[k];
  const float* op = obs + (size_t)k * NN + t0;     // 8-float aligned
  f4 o0 = *(const f4*)(op);
  f4 o1 = *(const f4*)(op + 4);
  float* p0 = lams0 + (size_t)k * NN + t0;
  float* p1 = lam1  + (size_t)k * NN + t0;
  float acc = 0.f;
#pragma unroll
  for (int r = 0; r < 8; ++r) {
    int t = t0 + r;
    float x = d[r];
    float l1 = (t == 0) ? 0.f : ((x > 20.f) ? x : log1pf(__expf(x)));
    float o = (r < 4) ? o0[r] : o1[r - 4];
    acc += o * __logf(mu + l1 + 1e-5f) - mu - l1;
    p0[r] = mu;      // scalar stores: d_out+1 base is only 4B aligned
    p1[r] = l1;
  }
  return acc;
}

#define LOAD_BFRAG(bb, bp, kk)                                                   \
  {                                                                              \
    h8 q0 = *(const h8*)((bp) + (kk) + kbb);                                     \
    h8 q1 = *(const h8*)((bp) + (kk) + kbb + 16);                                \
    bb = __builtin_shufflevector(q0, q1, 0,1,2,3,4,5,6,7,8,9,10,11,12,13,14,15); \
  }

__global__ __launch_bounds__(256)
void hawkes_gemm(const _Float16* __restrict__ G,      // [NN][KK]
                 const _Float16* __restrict__ AH,     // [KK][KK] (Alpha, row-major)
                 const float* __restrict__ obs,       // [KK][NN]
                 const float* __restrict__ Mu0,       // [KK]
                 float* __restrict__ out,             // [0]=loglik, +1: lams0, +1+KK*NN: lam1
                 float* __restrict__ blockSums) {
  int lane = threadIdx.x & 31;
  int wave = threadIdx.x >> 5;
  int wid  = blockIdx.x * 8 + wave;    // 8192 waves
  int tt   = wid >> 4;                 // 512 t-supertiles of 32 rows
  int kg   = wid & 15;                 // 16 groups of 4 k-tiles (64 channels)
  int tbase = tt * 32;

  int mrow = lane & 15;
  int kba  = (lane < 16) ? 0 : 8;      // A-fragment k offset (ISA 16-bit A 16x32 layout)
  int kbb  = (lane < 16) ? 0 : 8;      // B-fragment k offset (same k order as A fragment)
  int ncol = lane & 15;

  const _Float16* arow0 = G + (size_t)(tbase + mrow) * KK;
  const _Float16* arow1 = arow0 + (size_t)16 * KK;
  const _Float16* b0 = AH + (size_t)(kg * 64 +  0 + ncol) * KK;
  const _Float16* b1 = AH + (size_t)(kg * 64 + 16 + ncol) * KK;
  const _Float16* b2 = AH + (size_t)(kg * 64 + 32 + ncol) * KK;
  const _Float16* b3 = AH + (size_t)(kg * 64 + 48 + ncol) * KK;

  v8f acc00 = {}, acc01 = {}, acc02 = {}, acc03 = {};
  v8f acc10 = {}, acc11 = {}, acc12 = {}, acc13 = {};

  for (int kk = 0; kk < KK; kk += 32) {
    h8 alo = *(const h8*)(arow0 + kk + kba);
    h8 ahi = *(const h8*)(arow0 + kk + kba + 16);
    v16h a0 = __builtin_shufflevector(alo, ahi, 0,1,2,3,4,5,6,7,8,9,10,11,12,13,14,15);
    alo = *(const h8*)(arow1 + kk + kba);
    ahi = *(const h8*)(arow1 + kk + kba + 16);
    v16h a1 = __builtin_shufflevector(alo, ahi, 0,1,2,3,4,5,6,7,8,9,10,11,12,13,14,15);

    v16h bb;
    LOAD_BFRAG(bb, b0, kk);
    acc00 = __builtin_amdgcn_wmma_f32_16x16x32_f16(false, a0, false, bb, (short)0, acc00, false, false);
    acc10 = __builtin_amdgcn_wmma_f32_16x16x32_f16(false, a1, false, bb, (short)0, acc10, false, false);
    LOAD_BFRAG(bb, b1, kk);
    acc01 = __builtin_amdgcn_wmma_f32_16x16x32_f16(false, a0, false, bb, (short)0, acc01, false, false);
    acc11 = __builtin_amdgcn_wmma_f32_16x16x32_f16(false, a1, false, bb, (short)0, acc11, false, false);
    LOAD_BFRAG(bb, b2, kk);
    acc02 = __builtin_amdgcn_wmma_f32_16x16x32_f16(false, a0, false, bb, (short)0, acc02, false, false);
    acc12 = __builtin_amdgcn_wmma_f32_16x16x32_f16(false, a1, false, bb, (short)0, acc12, false, false);
    LOAD_BFRAG(bb, b3, kk);
    acc03 = __builtin_amdgcn_wmma_f32_16x16x32_f16(false, a0, false, bb, (short)0, acc03, false, false);
    acc13 = __builtin_amdgcn_wmma_f32_16x16x32_f16(false, a1, false, bb, (short)0, acc13, false, false);
  }

  // epilogue: per lane, acc[r] is 8 consecutive t values of one output channel
  int t0a = tbase + ((lane < 16) ? 0 : 8);
  int t0b = t0a + 16;
  float* lams0 = out + 1;
  float* lam1  = out + 1 + (size_t)KK * NN;
  float local = 0.f;
  local += epilogue_tile(acc00, kg * 64 +  0 + ncol, t0a, obs, Mu0, lams0, lam1);
  local += epilogue_tile(acc01, kg * 64 + 16 + ncol, t0a, obs, Mu0, lams0, lam1);
  local += epilogue_tile(acc02, kg * 64 + 32 + ncol, t0a, obs, Mu0, lams0, lam1);
  local += epilogue_tile(acc03, kg * 64 + 48 + ncol, t0a, obs, Mu0, lams0, lam1);
  local += epilogue_tile(acc10, kg * 64 +  0 + ncol, t0b, obs, Mu0, lams0, lam1);
  local += epilogue_tile(acc11, kg * 64 + 16 + ncol, t0b, obs, Mu0, lams0, lam1);
  local += epilogue_tile(acc12, kg * 64 + 32 + ncol, t0b, obs, Mu0, lams0, lam1);
  local += epilogue_tile(acc13, kg * 64 + 48 + ncol, t0b, obs, Mu0, lams0, lam1);

  __shared__ float red[256];
  red[threadIdx.x] = local; __syncthreads();
  for (int s = 128; s > 0; s >>= 1) {
    if (threadIdx.x < s) red[threadIdx.x] += red[threadIdx.x + s];
    __syncthreads();
  }
  if (threadIdx.x == 0) blockSums[blockIdx.x] = red[0];
}

// ---------------- deterministic final reduction of 1024 block sums ----------------
__global__ __launch_bounds__(256)
void finalred_kernel(const float* __restrict__ bs, float* __restrict__ out) {
  float s = 0.f;
  for (int i = threadIdx.x; i < 1024; i += 256) s += bs[i];
  __shared__ float red[256];
  red[threadIdx.x] = s; __syncthreads();
  for (int st = 128; st > 0; st >>= 1) {
    if (threadIdx.x < st) red[threadIdx.x] += red[threadIdx.x + st];
    __syncthreads();
  }
  if (threadIdx.x == 0) out[0] = red[0];
}

extern "C" void kernel_launch(void* const* d_in, const int* in_sizes, int n_in,
                              void* d_out, int out_size, void* d_ws, size_t ws_size,
                              hipStream_t stream) {
  const float* obs   = (const float*)d_in[0];   // [1024][16384]
  const float* Beta  = (const float*)d_in[1];   // [1024]
  const float* Alpha = (const float*)d_in[2];   // [1024][1024]
  float* out = (float*)d_out;
  char* ws = (char*)d_ws;

  // workspace layout (bytes)
  _Float16* G      = (_Float16*)(ws);                       // 16384*1024*2 = 33,554,432
  _Float16* AH     = (_Float16*)(ws + 33554432);            // 1024*1024*2  =  2,097,152
  float*    Mu0    = (float*)   (ws + 35651584);            // 1024*4       =      4,096
  float*    E      = (float*)   (ws + 35655680);            // 64*1024*4    =    262,144
  float*    Astart = (float*)   (ws + 35917824);            // 64*1024*4    =    262,144
  float*    bsum   = (float*)   (ws + 36179968);            // 1024*4       =      4,096

  mu0_kernel  <<<KK, 256, 0, stream>>>(obs, Mu0);
  a2h_kernel  <<<(KK * KK) / 256, 256, 0, stream>>>(Alpha, AH);
  scanA_kernel<<<(KK * NCH) / 256, 256, 0, stream>>>(obs, Beta, E);
  scanP_kernel<<<KK / 256, 256, 0, stream>>>(Beta, E, Astart);
  scanB_kernel<<<(KK * NCH) / 256, 256, 0, stream>>>(obs, Beta, Astart, G);
  hawkes_gemm <<<1024, 256, 0, stream>>>(G, AH, obs, Mu0, out, bsum);
  finalred_kernel<<<1, 256, 0, stream>>>(bsum, out);
}
